// HINGCN_GS_12137577579033
// MI455X (gfx1250) — matrix operand order repaired and checked by reading, not compiled, not run on hardware
//
#include <hip/hip_runtime.h>
#include <hip/hip_bf16.h>

// ---------------------------------------------------------------------------
// HINGCN-GS for MI455X (gfx1250): bf16 WMMA everywhere, fused mean-aggregation,
// deterministic LDS reductions, wave32 tiles (16x16 f32 accum per wave).
// Round 2: native bf16 cvt, b128 vector A loads, unrolled K loops.
// ---------------------------------------------------------------------------

typedef __bf16 bf16;
typedef __attribute__((ext_vector_type(16))) __bf16 v16bf;
typedef __attribute__((ext_vector_type(8)))  __bf16 v8bf;
typedef __attribute__((ext_vector_type(8)))  float  v8f;

#define BN   1024   // batch
#define S0N  10     // neighbor fanout (both hops)
#define PN   128    // input feature dim
#define EN   32     // edge dim
#define DN   256    // H*O concat dim
#define NCLS 8

// ----------------------------- device helpers ------------------------------

__device__ __forceinline__ bf16 f2bf(float x) { return (bf16)x; }  // RNE cvt

__device__ __forceinline__ v8f wmma_bf16(v16bf a, v16bf b, v8f c) {
    // emits v_wmma_f32_16x16x32_bf16
    return __builtin_amdgcn_wmma_f32_16x16x32_bf16(
        /*neg_a=*/false, a, /*neg_b=*/false, b,
        /*c_mod=*/(short)0, c, /*reuse_a=*/false, /*reuse_b=*/false);
}

// A operand (16x32 bf16, M x K): lane m=lane&15, hi=lane>>4 holds
// K = hi*8 + {0..7} (VGPRs 0-3) and K = hi*8 + {16..23} (VGPRs 4-7):
// two contiguous 16B runs -> two global_load_b128 per chunk.
__device__ __forceinline__ v16bf load_a_row_bf16(const bf16* rowp, int k0, int hi) {
    const bf16* base = rowp + k0 + hi * 8;
    v8bf lo = *(const v8bf*)(base);
    v8bf hh = *(const v8bf*)(base + 16);
    return __builtin_shufflevector(lo, hh, 0, 1, 2, 3, 4, 5, 6, 7,
                                           8, 9, 10, 11, 12, 13, 14, 15);
}

// Same A layout from gathered f32 rows: four float4 loads + native cvt.
__device__ __forceinline__ v16bf load_a_row_f32(const float* rowp, int k0, int hi) {
    const float* base = rowp + k0 + hi * 8;
    float4 p0 = *(const float4*)(base + 0);
    float4 p1 = *(const float4*)(base + 4);
    float4 p2 = *(const float4*)(base + 16);
    float4 p3 = *(const float4*)(base + 20);
    v16bf a;
    a[0]  = f2bf(p0.x); a[1]  = f2bf(p0.y); a[2]  = f2bf(p0.z); a[3]  = f2bf(p0.w);
    a[4]  = f2bf(p1.x); a[5]  = f2bf(p1.y); a[6]  = f2bf(p1.z); a[7]  = f2bf(p1.w);
    a[8]  = f2bf(p2.x); a[9]  = f2bf(p2.y); a[10] = f2bf(p2.z); a[11] = f2bf(p2.w);
    a[12] = f2bf(p3.x); a[13] = f2bf(p3.y); a[14] = f2bf(p3.z); a[15] = f2bf(p3.w);
    return a;
}

// B operand (32x16 bf16, K x N), pre-packed: tile-major [(kt*NT+ct)*32 + lane]
// contiguous v16bf per lane -> one 32B load.
__device__ __forceinline__ v16bf load_b(const bf16* Wp, int kt, int ct, int NT, int lane) {
    return ((const v16bf*)Wp)[(size_t)(kt * NT + ct) * 32 + lane];
}

// ------------------------------- kernels -----------------------------------

// Pack a K x N f32 weight matrix into the WMMA-B-native bf16 layout.
// headFlat=1 : source is W[H=8][K][O=32] with N = h*32+o (head-concat flatten).
__global__ __launch_bounds__(256) void pack_weight_kernel(
        const float* __restrict__ src, bf16* __restrict__ dst,
        int K, int N, int headFlat) {
    int tid = blockIdx.x * blockDim.x + threadIdx.x;
    if (tid >= K * N) return;
    int j    = tid & 15;
    int lane = (tid >> 4) & 31;
    int tile = tid >> 9;
    int NT = N >> 4;
    int ct = tile % NT, kt = tile / NT;
    int n  = lane & 15, hi = lane >> 4;
    int v = j >> 1, slot = j & 1;
    int gk = kt * 32 + ((2 * v + slot) & 7) + hi * 8 + ((2 * v + slot) >> 3) * 16;   // B layout: K = hi*8 + {0..7}, hi*8 + {16..23}
    int gn = ct * 16 + n;
    size_t off = headFlat ? (((size_t)(gn >> 5) * K + gk) * 32 + (gn & 31))
                          : ((size_t)gk * N + gn);
    dst[tid] = f2bf(src[off]);
}

// relu(gather(feats, idx) @ Wprep) -> bf16 [M,128]. One 16x16 tile per wave.
__global__ __launch_bounds__(256) void prep_kernel(
        const float* __restrict__ feats, const int* __restrict__ idx,
        const bf16* __restrict__ WprepP, bf16* __restrict__ out, int M) {
    int wave = (blockIdx.x * blockDim.x + threadIdx.x) >> 5;
    int lane = threadIdx.x & 31;
    int tiles = (M >> 4) * 8;                    // 8 col tiles (N=128)
    if (wave >= tiles) return;
    int rt = wave >> 3, ct = wave & 7;
    int row0 = rt * 16;
    int m = lane & 15, hi = lane >> 4, n = m;
    const float* rowp = feats + (size_t)idx[row0 + m] * PN;
    v8f c = {};
#pragma unroll
    for (int kt = 0; kt < 4; ++kt) {             // K = 128
        v16bf a = load_a_row_f32(rowp, kt * 32, hi);
        v16bf b = load_b(WprepP, kt, ct, 8, lane);
        c = wmma_bf16(a, b, c);
    }
#pragma unroll
    for (int r = 0; r < 8; ++r) {
        float v = fmaxf(c[r], 0.f);
        out[(size_t)(row0 + r + 8 * hi) * PN + ct * 16 + n] = f2bf(v);
    }
}

// ed[r, 0:32] = bf16(edge_emb[e[r]])
__global__ __launch_bounds__(256) void gather_edges_kernel(
        const float* __restrict__ emb, const int* __restrict__ e,
        bf16* __restrict__ out, int total) {
    int tid = blockIdx.x * blockDim.x + threadIdx.x;
    if (tid >= total) return;
    out[tid] = f2bf(emb[(size_t)e[tid >> 5] * EN + (tid & 31)]);
}

// Fused: msg = relu(neigh @ WnP + ed @ WeP); agg = mean over groups of 10.
// Block = 16 output groups (160 msg rows) x 16 output cols. 10 row-tiles over
// 8 waves; relu'd tiles staged in LDS; deterministic group reduction.
template<int KN>
__global__ __launch_bounds__(256) void msg_mean_kernel(
        const bf16* __restrict__ neigh, const bf16* __restrict__ WnP,
        const bf16* __restrict__ ed, const bf16* __restrict__ WeP,
        float* __restrict__ agg, int numGroups) {
    __shared__ float msgT[10 * 16 * 16];         // 10 KB of 320 KB/WGP
    int tid = threadIdx.x;
    int waveId = tid >> 5, lane = tid & 31;
    int ct = blockIdx.y;
    int groupBase = blockIdx.x * 16;
    int rowBase = groupBase * S0N;
    int m = lane & 15, hi = lane >> 4, n = m;
    for (int t = waveId; t < 10; t += 8) {
        int row0 = rowBase + t * 16;
        const bf16* nrow = neigh + (size_t)(row0 + m) * KN;
        const bf16* erow = ed    + (size_t)(row0 + m) * EN;
        v8f c = {};
#pragma unroll
        for (int kt = 0; kt < KN / 32; ++kt) {   // neighbor GEMM (unrolled)
            v16bf a = load_a_row_bf16(nrow, kt * 32, hi);
            v16bf b = load_b(WnP, kt, ct, 16, lane);
            c = wmma_bf16(a, b, c);
        }
        {                                        // edge GEMM (K=32, one chunk)
            v16bf a = load_a_row_bf16(erow, 0, hi);
            v16bf b = load_b(WeP, 0, ct, 16, lane);
            c = wmma_bf16(a, b, c);
        }
#pragma unroll
        for (int r = 0; r < 8; ++r)
            msgT[(t * 16 + r + 8 * hi) * 16 + n] = fmaxf(c[r], 0.f);
    }
    __syncthreads();
    {   // 256 threads -> (group g, col nn); deterministic sum of 10 rows
        int g = tid >> 4, nn = tid & 15;
        float s = 0.f;
#pragma unroll
        for (int i = 0; i < 10; ++i)
            s += msgT[(g * S0N + i) * 16 + nn];
        agg[(size_t)(groupBase + g) * DN + ct * 16 + nn] = s * 0.1f;
    }
}

// out = relu(x @ WsP + agg); agg is loaded straight into the WMMA C operand.
// Stores bf16 (for downstream GEMMs) and/or f32 (final layer feeds attention).
template<int KD>
__global__ __launch_bounds__(256) void self_kernel(
        const bf16* __restrict__ x, const bf16* __restrict__ WsP,
        const float* __restrict__ agg, bf16* __restrict__ outB,
        float* __restrict__ outF, int M) {
    int wave = (blockIdx.x * blockDim.x + threadIdx.x) >> 5;
    int lane = threadIdx.x & 31;
    int tiles = (M >> 4) * 16;                   // N = 256 -> 16 col tiles
    if (wave >= tiles) return;
    int rt = wave >> 4, ct = wave & 15;
    int row0 = rt * 16;
    int m = lane & 15, hi = lane >> 4, n = m;
    const bf16* xrow = x + (size_t)(row0 + m) * KD;
    v8f c;
#pragma unroll
    for (int r = 0; r < 8; ++r)                  // C init = mean-aggregate
        c[r] = agg[(size_t)(row0 + r + 8 * hi) * DN + ct * 16 + n];
#pragma unroll
    for (int kt = 0; kt < KD / 32; ++kt) {
        v16bf a = load_a_row_bf16(xrow, kt * 32, hi);
        v16bf b = load_b(WsP, kt, ct, 16, lane);
        c = wmma_bf16(a, b, c);
    }
#pragma unroll
    for (int r = 0; r < 8; ++r) {
        float v = fmaxf(c[r], 0.f);
        size_t o = (size_t)(row0 + r + 8 * hi) * DN + ct * 16 + n;
        if (outB) outB[o] = f2bf(v);
        if (outF) outF[o] = v;
    }
}

// ed0u = relu(concat([g0[r/10], g1[r], ed0[r]]) @ Wedge0) without materializing
// the 544-wide concat: three accumulating K-loops against row-slices of WedP.
__global__ __launch_bounds__(256) void edge_update_kernel(
        const bf16* __restrict__ g0, const bf16* __restrict__ g1,
        const bf16* __restrict__ ed0, const bf16* __restrict__ WP,
        bf16* __restrict__ out, int M) {
    int wave = (blockIdx.x * blockDim.x + threadIdx.x) >> 5;
    int lane = threadIdx.x & 31;
    int tiles = (M >> 4) * 2;                    // N = 32 -> 2 col tiles
    if (wave >= tiles) return;
    int rt = wave >> 1, ct = wave & 1;
    int row0 = rt * 16;
    int m = lane & 15, hi = lane >> 4, n = m;
    const bf16* srow = g0  + (size_t)((row0 + m) / S0N) * DN;  // repeat(g0, 10)
    const bf16* grow = g1  + (size_t)(row0 + m) * DN;
    const bf16* erow = ed0 + (size_t)(row0 + m) * EN;
    v8f c = {};
#pragma unroll
    for (int kt = 0; kt < 8; ++kt) {             // W rows   0..255 : src
        v16bf a = load_a_row_bf16(srow, kt * 32, hi);
        c = wmma_bf16(a, load_b(WP, kt, ct, 2, lane), c);
    }
#pragma unroll
    for (int kt = 0; kt < 8; ++kt) {             // W rows 256..511 : g1
        v16bf a = load_a_row_bf16(grow, kt * 32, hi);
        c = wmma_bf16(a, load_b(WP, 8 + kt, ct, 2, lane), c);
    }
    {                                            // W rows 512..543 : ed0
        v16bf a = load_a_row_bf16(erow, 0, hi);
        c = wmma_bf16(a, load_b(WP, 16, ct, 2, lane), c);
    }
#pragma unroll
    for (int r = 0; r < 8; ++r) {
        float v = fmaxf(c[r], 0.f);
        out[(size_t)(row0 + r + 8 * hi) * EN + ct * 16 + n] = f2bf(v);
    }
}

// Metapath attention + softmax + L2 normalize + classifier. One wave per row.
__global__ __launch_bounds__(256) void attention_kernel(
        const float* __restrict__ outF, const float* __restrict__ Wa,
        const float* __restrict__ va, const float* __restrict__ Wfc,
        const float* __restrict__ bfc, float* __restrict__ logits,
        float* __restrict__ wts) {
    int wave = (blockIdx.x * blockDim.x + threadIdx.x) >> 5;
    int lane = threadIdx.x & 31;
    if (wave >= BN) return;
    int b = wave;
    float s[2];
    for (int mp = 0; mp < 2; ++mp) {
        const float* o = outF + ((size_t)mp * BN + b) * DN;
        float acc = 0.f;
#pragma unroll
        for (int jj = 0; jj < 2; ++jj) {         // 64 cols over 32 lanes
            int j = lane + 32 * jj;
            float dt = 0.f;
            for (int d = 0; d < DN; ++d) dt += o[d] * Wa[d * 64 + j];
            acc += va[j] * tanhf(dt);
        }
        for (int off = 16; off; off >>= 1) acc += __shfl_xor(acc, off, 32);
        s[mp] = acc;
    }
    float mx = fmaxf(s[0], s[1]);
    float e0 = __expf(s[0] - mx), e1 = __expf(s[1] - mx);
    float w0 = e0 / (e0 + e1), w1 = e1 / (e0 + e1);
    if (lane == 0) { wts[b] = w0; wts[BN + b] = w1; }
    const float* o0 = outF + (size_t)b * DN;
    const float* o1 = outF + (size_t)(BN + b) * DN;
    float ad[8], ss = 0.f;
#pragma unroll
    for (int i = 0; i < 8; ++i) {
        int d = lane + 32 * i;
        ad[i] = w0 * o0[d] + w1 * o1[d];
        ss += ad[i] * ad[i];
    }
    for (int off = 16; off; off >>= 1) ss += __shfl_xor(ss, off, 32);
    float inv = 1.f / fmaxf(sqrtf(ss), 1e-12f);
    float lc[NCLS];
#pragma unroll
    for (int c = 0; c < NCLS; ++c) lc[c] = 0.f;
#pragma unroll
    for (int i = 0; i < 8; ++i) {
        int d = lane + 32 * i;
        float nv = ad[i] * inv;
#pragma unroll
        for (int c = 0; c < NCLS; ++c) lc[c] += nv * Wfc[d * NCLS + c];
    }
#pragma unroll
    for (int c = 0; c < NCLS; ++c) {
        float v = lc[c];
        for (int off = 16; off; off >>= 1) v += __shfl_xor(v, off, 32);
        if (lane == 0) logits[b * NCLS + c] = v + bfc[c];
    }
}

// ------------------------------ host launch --------------------------------

extern "C" void kernel_launch(void* const* d_in, const int* in_sizes, int n_in,
                              void* d_out, int out_size, void* d_ws, size_t ws_size,
                              hipStream_t stream) {
    (void)in_sizes; (void)n_in; (void)out_size; (void)ws_size;
    const int*   ids       = (const int*)d_in[0];
    const int*   n0[2]     = {(const int*)d_in[1], (const int*)d_in[5]};
    const int*   e0[2]     = {(const int*)d_in[2], (const int*)d_in[6]};
    const int*   n1[2]     = {(const int*)d_in[3], (const int*)d_in[7]};
    const int*   e1[2]     = {(const int*)d_in[4], (const int*)d_in[8]};
    const float* feats     = (const float*)d_in[9];
    const float* edge_emb[2] = {(const float*)d_in[10], (const float*)d_in[11]};
    const float* Wprep = (const float*)d_in[12];
    const float* Ws0   = (const float*)d_in[13];
    const float* Wn0   = (const float*)d_in[14];
    const float* We0   = (const float*)d_in[15];
    const float* Ws1   = (const float*)d_in[16];
    const float* Wn1   = (const float*)d_in[17];
    const float* We1   = (const float*)d_in[18];
    const float* Wedge0= (const float*)d_in[19];
    const float* Wa    = (const float*)d_in[20];
    const float* va    = (const float*)d_in[21];
    const float* Wfc   = (const float*)d_in[22];
    const float* bfc   = (const float*)d_in[23];
    float* logits = (float*)d_out;               // [1024,8]
    float* wts    = (float*)d_out + BN * NCLS;   // [2,1024]

    // Workspace carve-up (~56 MB total), 256B aligned for 32B vector B loads.
    char* wp = (char*)d_ws;
    auto alloc = [&](size_t bytes) -> void* {
        void* p = (void*)wp;
        wp += (bytes + 255) & ~(size_t)255;
        return p;
    };
    bf16* WprepP = (bf16*)alloc((size_t)128 * 128 * 2);
    bf16* Ws0P   = (bf16*)alloc((size_t)2 * 128 * 256 * 2);
    bf16* Wn0P   = (bf16*)alloc((size_t)2 * 128 * 256 * 2);
    bf16* We0P   = (bf16*)alloc((size_t)2 * 32  * 256 * 2);
    bf16* Ws1P   = (bf16*)alloc((size_t)2 * 256 * 256 * 2);
    bf16* Wn1P   = (bf16*)alloc((size_t)2 * 256 * 256 * 2);
    bf16* We1P   = (bf16*)alloc((size_t)2 * 32  * 256 * 2);
    bf16* WedP   = (bf16*)alloc((size_t)2 * 544 * 32  * 2);
    bf16* f0   = (bf16*)alloc((size_t)BN * PN * 2);
    bf16* f1   = (bf16*)alloc((size_t)BN * S0N * PN * 2);
    bf16* f2   = (bf16*)alloc((size_t)BN * S0N * S0N * PN * 2);
    bf16* ed0  = (bf16*)alloc((size_t)BN * S0N * EN * 2);
    bf16* ed1  = (bf16*)alloc((size_t)BN * S0N * S0N * EN * 2);
    float* agg = (float*)alloc((size_t)BN * S0N * DN * 4);
    bf16* g0   = (bf16*)alloc((size_t)BN * DN * 2);
    bf16* g1   = (bf16*)alloc((size_t)BN * S0N * DN * 2);
    bf16* ed0u = (bf16*)alloc((size_t)BN * S0N * EN * 2);
    float* outF= (float*)alloc((size_t)2 * BN * DN * 4);

    auto pack = [&](const float* src, bf16* dst, int K, int N, int hf) {
        int total = K * N;
        pack_weight_kernel<<<(total + 255) / 256, 256, 0, stream>>>(src, dst, K, N, hf);
    };
    pack(Wprep, WprepP, 128, 128, 0);
    for (int mp = 0; mp < 2; ++mp) {
        pack(Ws0 + (size_t)mp * 8 * 128 * 32, Ws0P + (size_t)mp * 128 * 256, 128, 256, 1);
        pack(Wn0 + (size_t)mp * 8 * 128 * 32, Wn0P + (size_t)mp * 128 * 256, 128, 256, 1);
        pack(We0 + (size_t)mp * 8 * 32  * 32, We0P + (size_t)mp * 32  * 256, 32,  256, 1);
        pack(Ws1 + (size_t)mp * 8 * 256 * 32, Ws1P + (size_t)mp * 256 * 256, 256, 256, 1);
        pack(Wn1 + (size_t)mp * 8 * 256 * 32, Wn1P + (size_t)mp * 256 * 256, 256, 256, 1);
        pack(We1 + (size_t)mp * 8 * 32  * 32, We1P + (size_t)mp * 32  * 256, 32,  256, 1);
        pack(Wedge0 + (size_t)mp * 544 * 32,  WedP + (size_t)mp * 544 * 32,  544, 32,  0);
    }

    // f0 = prep(ids) is shared by both metapaths (same ids, shared Wprep).
    prep_kernel<<<BN / 16, 256, 0, stream>>>(feats, ids, WprepP, f0, BN);

    for (int mp = 0; mp < 2; ++mp) {
        const bf16* Ws0m = Ws0P + (size_t)mp * 128 * 256;
        const bf16* Wn0m = Wn0P + (size_t)mp * 128 * 256;
        const bf16* We0m = We0P + (size_t)mp * 32  * 256;
        const bf16* Ws1m = Ws1P + (size_t)mp * 256 * 256;
        const bf16* Wn1m = Wn1P + (size_t)mp * 256 * 256;
        const bf16* We1m = We1P + (size_t)mp * 32  * 256;
        const bf16* Wedm = WedP + (size_t)mp * 544 * 32;
        const int M1 = BN * S0N, M2 = BN * S0N * S0N;

        prep_kernel<<<M1 / 16, 256, 0, stream>>>(feats, n0[mp], WprepP, f1, M1);
        prep_kernel<<<M2 / 16, 256, 0, stream>>>(feats, n1[mp], WprepP, f2, M2);
        gather_edges_kernel<<<(M1 * EN + 255) / 256, 256, 0, stream>>>(edge_emb[mp], e0[mp], ed0, M1 * EN);
        gather_edges_kernel<<<(M2 * EN + 255) / 256, 256, 0, stream>>>(edge_emb[mp], e1[mp], ed1, M2 * EN);

        // layer-0 aggregator on hop pair (f0, f1, ed0) -> g0 [B, 256]
        msg_mean_kernel<128><<<dim3(BN / 16, 16), 256, 0, stream>>>(f1, Wn0m, ed0, We0m, agg, BN);
        self_kernel<128><<<(BN + 7) / 8, 256, 0, stream>>>(f0, Ws0m, agg, g0, nullptr, BN);

        // layer-0 aggregator on hop pair (f1, f2, ed1) -> g1 [B*S0, 256]
        msg_mean_kernel<128><<<dim3(M1 / 16, 16), 256, 0, stream>>>(f2, Wn0m, ed1, We0m, agg, M1);
        self_kernel<128><<<(M1 + 7) / 8, 256, 0, stream>>>(f1, Ws0m, agg, g1, nullptr, M1);

        // edge update on hop-0 edges -> ed0u [B*S0, 32]
        edge_update_kernel<<<((M1 / 16) * 2 + 7) / 8, 256, 0, stream>>>(g0, g1, ed0, Wedm, ed0u, M1);

        // layer-1 aggregator (g0, g1, ed0u) -> outF[mp] [B, 256] (f32)
        msg_mean_kernel<256><<<dim3(BN / 16, 16), 256, 0, stream>>>(g1, Wn1m, ed0u, We1m, agg, BN);
        self_kernel<256><<<(BN + 7) / 8, 256, 0, stream>>>(g0, Ws1m, agg, nullptr,
                                                           outF + (size_t)mp * BN * DN, BN);
    }

    attention_kernel<<<BN / 8, 256, 0, stream>>>(outF, Wa, va, Wfc, bfc, logits, wts);
}
